// SwinTransformerBlock_48387101556861
// MI455X (gfx1250) — matrix operand order repaired and checked by reading, not compiled, not run on hardware
//
#include <hip/hip_runtime.h>
#include <hip/hip_bf16.h>
#include <math.h>
#include <stdint.h>

// ---------------------------------------------------------------------------
// Swin Transformer block (B=32, 56x56, C=384, NH=12, WS=7, shift=3, MLP=1536)
// - all GEMMs on v_wmma_f32_16x16x32_bf16 (wave32)
// - LDS fragment reads are 16B ds_load_b128 (pre-transposed bf16 weights)
// - tile staging uses gfx1250 async copy global_load_async_to_lds_b128 with
//   double-buffered LDS tiles (prefetch of K+1 overlaps WMMA on K)
// ---------------------------------------------------------------------------

#define DEVI __device__ __forceinline__

typedef __attribute__((ext_vector_type(16))) __bf16 v16bf;
typedef __attribute__((ext_vector_type(8)))  __bf16 v8bf;
typedef __attribute__((ext_vector_type(8)))  float  v8f;

namespace cfg {
constexpr int B     = 32;
constexpr int HS    = 56;
constexpr int WSS   = 56;
constexpr int C     = 384;
constexpr int NH    = 12;
constexpr int HD    = 32;       // C / NH
constexpr int WS    = 7;
constexpr int SHIFT = 3;
constexpr int MLP   = 1536;
constexpr int N     = 49;       // WS*WS tokens per window
constexpr int NWIN  = 2048;     // B * 8 * 8 windows
constexpr int TOK   = 100352;   // B * HS * WSS
constexpr int TILE  = 2560;     // 64 rows * ld 40, one LDS tile buffer
}

// --------------------------- async copy helpers ----------------------------
// LDS aperture occupies addr[63:32]; addr[31:0] is the LDS byte offset, so a
// 32-bit truncation of a flat shared pointer is the DS address.
DEVI void async_wait0() {
    asm volatile("s_wait_asynccnt 0x0" ::: "memory");
}

DEVI void async_b128(__bf16* lds_dst, const __bf16* gsrc) {
    unsigned lds = (unsigned)(uintptr_t)lds_dst;
    unsigned long long g = (unsigned long long)(uintptr_t)gsrc;
    asm volatile("global_load_async_to_lds_b128 %0, %1, off"
                 :: "v"(lds), "v"(g) : "memory");
}

// Issue async loads for a 64x32 bf16 tile (row-major, src_ld elems) into an
// LDS tile with ld=40. Only the first ROWS rows are fetched.
template<int ROWS>
DEVI void stage_async(__bf16* dst, const __bf16* src, int src_ld, int tid) {
#pragma unroll
    for (int base = 0; base < ROWS * 4; base += 128) {
        int idx = base + tid;
        if (ROWS * 4 - base >= 128 || idx < ROWS * 4) {
            int r  = idx >> 2;
            int c8 = (idx & 3) << 3;
            async_b128(dst + r * 40 + c8, src + (size_t)r * src_ld + c8);
        }
    }
}

// --------------------------- WMMA fragment helpers -------------------------
// A-matrix 16x32 bf16 (MxK): lane row=(lane&15); kg=8*(lane>>4); two
// contiguous 8-elem runs (K kg.. and 16+kg..) -> two 16B LDS loads.
DEVI v16bf frag_a(const __bf16* s, int ld, int row0, int k0) {
    int lane = threadIdx.x & 31;
    int r  = row0 + (lane & 15);
    int kg = (lane >> 4) << 3;
    const __bf16* p = s + r * ld + k0 + kg;
    union { v16bf v; v8bf h[2]; } u;
    u.h[0] = *(const v8bf*)(p);
    u.h[1] = *(const v8bf*)(p + 16);
    return u.v;
}

// B-matrix 32x16 (KxN) fragment gathered from an [n][k] row-major tile
// (B^T stored row-major): two 8-elem K runs (kg.. and 16+kg..), kg=8*(lane>>4).
DEVI v16bf frag_bT(const __bf16* s, int ld, int k0, int col0) {
    int lane = threadIdx.x & 31;
    int n  = col0 + (lane & 15);
    int kk = k0 + ((lane >> 4) << 3);
    const __bf16* p = s + n * ld + kk;
    union { v16bf v; v8bf h[2]; } u;
    u.h[0] = *(const v8bf*)(p);
    u.h[1] = *(const v8bf*)(p + 16);
    return u.v;
}

DEVI v8f wmma_bf16(v16bf a, v16bf b, v8f c) {
    return __builtin_amdgcn_wmma_f32_16x16x32_bf16(
        false, a, false, b, (short)0, c, false, false);
}

// C/D 16x16 f32 layout: lane col=(lane&15), VGPR i -> row i + 8*(lane>>4).

// Double-buffered GEMM main loop: acc[4] covers rows [16wv,16wv+16) x 64 cols.
// As/Bs are 2*TILE LDS buffers. A rows >= AROWS are zero (padded once).
template<int KDIM, int AROWS>
DEVI void gemm_mainloop(__bf16* As, __bf16* Bs,
                        const __bf16* Aw, int lda,
                        const __bf16* Bw, int ldb,
                        int tid, int wv, v8f* acc) {
    if (AROWS < 64) {
        constexpr int PAD = (64 - AROWS) * 4;   // 16B chunks per buffer
        for (int idx = tid; idx < 2 * PAD; idx += 128) {
            int b  = idx / PAD;
            int rr = idx - b * PAD;
            int r  = AROWS + (rr >> 2);
            int c8 = (rr & 3) << 3;
            v8bf z = {};
            *(v8bf*)(As + b * cfg::TILE + r * 40 + c8) = z;
        }
    }
    stage_async<AROWS>(As, Aw, lda, tid);
    stage_async<64>(Bs, Bw, ldb, tid);
    async_wait0();
    __syncthreads();

    int buf = 0;
    for (int k0 = 0; k0 < KDIM; k0 += 32, buf ^= 1) {
        if (k0 + 32 < KDIM) {   // prefetch next K-slice into the other buffer
            stage_async<AROWS>(As + (buf ^ 1) * cfg::TILE, Aw + k0 + 32, lda, tid);
            stage_async<64>(Bs + (buf ^ 1) * cfg::TILE, Bw + k0 + 32, ldb, tid);
        }
        const __bf16* Ab = As + buf * cfg::TILE;
        const __bf16* Bb = Bs + buf * cfg::TILE;
        v16bf a = frag_a(Ab, 40, wv * 16, 0);
#pragma unroll
        for (int t = 0; t < 4; ++t) {
            v16bf b = frag_bT(Bb, 40, 0, t * 16);
            acc[t] = wmma_bf16(a, b, acc[t]);
        }
        async_wait0();
        __syncthreads();
    }
}

// --------------------------- small utility kernels -------------------------

// d[n*K + k] = (bf16) s[k*Nn + n]  -- fused fp32->bf16 convert + transpose
__global__ void cvt_transpose_kernel(const float* __restrict__ s,
                                     __bf16* __restrict__ d, int K, int Nn) {
    int idx = blockIdx.x * 256 + threadIdx.x;
    if (idx < K * Nn) {
        int n = idx / K, k = idx - n * K;
        d[idx] = (__bf16)s[(size_t)k * Nn + n];
    }
}

// LayerNorm over C=384, one wave per token (12 elems/lane), fused with the
// cyclic shift (-3,-3) + window partition, bf16 output.
__global__ __launch_bounds__(256)
void ln1_window_kernel(const float* __restrict__ x,
                       const float* __restrict__ w,
                       const float* __restrict__ b,
                       __bf16* __restrict__ xw) {
    int lane = threadIdx.x & 31;
    int wv   = threadIdx.x >> 5;
    int tok  = blockIdx.x * 8 + wv;          // window-token id, < 100352
    int win  = tok / cfg::N;
    int n    = tok - win * cfg::N;
    int bidx = win >> 6;
    int wrem = win & 63;
    int wi = wrem >> 3, wj = wrem & 7;
    int hh = wi * cfg::WS + n / cfg::WS;
    int ww = wj * cfg::WS + n % cfg::WS;
    int sh = (hh + cfg::SHIFT) % cfg::HS;
    int sw = (ww + cfg::SHIFT) % cfg::WSS;
    const float* row = x + ((size_t)bidx * 3136 + sh * 56 + sw) * cfg::C;

    float vals[12];
    float s = 0.f;
#pragma unroll
    for (int j = 0; j < 12; ++j) { vals[j] = row[lane + 32 * j]; s += vals[j]; }
#pragma unroll
    for (int d = 1; d < 32; d <<= 1) s += __shfl_xor(s, d, 32);
    float mean = s * (1.f / cfg::C);
    float vs = 0.f;
#pragma unroll
    for (int j = 0; j < 12; ++j) { float dd = vals[j] - mean; vs += dd * dd; }
#pragma unroll
    for (int d = 1; d < 32; d <<= 1) vs += __shfl_xor(vs, d, 32);
    float inv = rsqrtf(vs * (1.f / cfg::C) + 1e-5f);

    __bf16* o = xw + (size_t)tok * cfg::C;
#pragma unroll
    for (int j = 0; j < 12; ++j) {
        int c = lane + 32 * j;
        o[c] = (__bf16)((vals[j] - mean) * inv * w[c] + b[c]);
    }
}

// Plain LayerNorm (post-attention residual), bf16 output.
__global__ __launch_bounds__(256)
void ln2_kernel(const float* __restrict__ xa,
                const float* __restrict__ w,
                const float* __restrict__ b,
                __bf16* __restrict__ hln) {
    int lane = threadIdx.x & 31;
    int wv   = threadIdx.x >> 5;
    int tok  = blockIdx.x * 8 + wv;
    const float* row = xa + (size_t)tok * cfg::C;

    float vals[12];
    float s = 0.f;
#pragma unroll
    for (int j = 0; j < 12; ++j) { vals[j] = row[lane + 32 * j]; s += vals[j]; }
#pragma unroll
    for (int d = 1; d < 32; d <<= 1) s += __shfl_xor(s, d, 32);
    float mean = s * (1.f / cfg::C);
    float vs = 0.f;
#pragma unroll
    for (int j = 0; j < 12; ++j) { float dd = vals[j] - mean; vs += dd * dd; }
#pragma unroll
    for (int d = 1; d < 32; d <<= 1) vs += __shfl_xor(vs, d, 32);
    float inv = rsqrtf(vs * (1.f / cfg::C) + 1e-5f);

    __bf16* o = hln + (size_t)tok * cfg::C;
#pragma unroll
    for (int j = 0; j < 12; ++j) {
        int c = lane + 32 * j;
        o[c] = (__bf16)((vals[j] - mean) * inv * w[c] + b[c]);
    }
}

// --------------------------- GEMM kernels (WMMA) ---------------------------

// qkv: xw(2048x49x384) @ qkv_w(384x1152) + b  -> qkvbuf bf16 (2048x49x1152)
__global__ __launch_bounds__(128)
void qkv_gemm_kernel(const __bf16* __restrict__ xw,
                     const __bf16* __restrict__ wqT,   // (1152 x 384)
                     const float* __restrict__ qb,
                     __bf16* __restrict__ qkvbuf) {
    int win   = blockIdx.y;
    int ncol0 = blockIdx.x * 64;
    __shared__ __align__(16) __bf16 As[2 * cfg::TILE];
    __shared__ __align__(16) __bf16 Bs[2 * cfg::TILE];
    int tid = threadIdx.x, wv = tid >> 5, lane = tid & 31;
    int half = lane >> 4, cl = lane & 15;
    v8f acc[4] = {};
    gemm_mainloop<cfg::C, cfg::N>(As, Bs,
                                  xw + (size_t)win * cfg::N * cfg::C, cfg::C,
                                  wqT + (size_t)ncol0 * cfg::C, cfg::C,
                                  tid, wv, acc);
#pragma unroll
    for (int t = 0; t < 4; ++t)
#pragma unroll
        for (int i = 0; i < 8; ++i) {
            int r = wv * 16 + 8 * half + i;
            if (r < cfg::N) {
                int col = ncol0 + t * 16 + cl;
                qkvbuf[(size_t)win * cfg::N * 1152 + (size_t)r * 1152 + col] =
                    (__bf16)(acc[t][i] + qb[col]);
            }
        }
}

// Attention per (head, window): S = (Q K^T)*scale + rpb + mask; softmax; O=P V
__global__ __launch_bounds__(128)
void attn_kernel(const __bf16* __restrict__ qkvbuf,
                 const float* __restrict__ rpb,     // (169, 12)
                 const int*   __restrict__ rpi,     // (49, 49)
                 const float* __restrict__ maskm,   // (64, 49, 49)
                 __bf16* __restrict__ aout) {       // (2048, 49, 384)
    int head = blockIdx.x;
    int win  = blockIdx.y;
    __shared__ __align__(16) __bf16 Qs[64 * 40];   // [token][hd]
    __shared__ __align__(16) __bf16 Ks[64 * 40];   // [token][hd]
    __shared__ __align__(16) __bf16 Vt[32 * 72];   // [hd][token]  (V^T)
    __shared__ __align__(16) __bf16 Ps[64 * 72];   // [row][token]
    int tid = threadIdx.x, wv = tid >> 5, lane = tid & 31;
    int half = lane >> 4, cl = lane & 15;

    const __bf16* base = qkvbuf + (size_t)win * cfg::N * 1152;

    // zero-pad rows 49..63 of Qs/Ks (60 chunks each, 120 total)
    if (tid < 120) {
        int which = tid >= 60;
        int rr = tid - which * 60;
        int r  = cfg::N + (rr >> 2);
        int c8 = (rr & 3) << 3;
        v8bf z = {};
        *(v8bf*)((which ? Ks : Qs) + r * 40 + c8) = z;
    }
    // async-stage Q and K rows [0,49)
    stage_async<cfg::N>(Qs, base + head * cfg::HD, 1152, tid);
    stage_async<cfg::N>(Ks, base + 384 + head * cfg::HD, 1152, tid);
    // V staged transposed (register transpose, scalar DS stores)
    for (int idx = tid; idx < 256; idx += 128) {
        int r  = idx >> 2;
        int c8 = (idx & 3) << 3;
        v8bf v = {};
        if (r < cfg::N)
            v = *(const v8bf*)(base + (size_t)r * 1152 + 768 + head * cfg::HD + c8);
#pragma unroll
        for (int j = 0; j < 8; ++j) Vt[(c8 + j) * 72 + r] = v[j];
    }
    async_wait0();
    __syncthreads();

    // S strip: rows [16wv,16wv+16) x 64 cols, K = hd = 32 (single WMMA step)
    v8f s[4];
    {
        v16bf a = frag_a(Qs, 40, wv * 16, 0);
#pragma unroll
        for (int t = 0; t < 4; ++t) {
            v16bf b = frag_bT(Ks, 40, 0, t * 16);
            v8f z = {};
            s[t] = wmma_bf16(a, b, z);
        }
    }

    const float scale = 0.17677669529663687f;  // hd^-0.5
    const float* mk = maskm + (size_t)(win & 63) * cfg::N * cfg::N;
#pragma unroll
    for (int t = 0; t < 4; ++t)
#pragma unroll
        for (int i = 0; i < 8; ++i) {
            int r = wv * 16 + 8 * half + i;
            int m = t * 16 + cl;
            float lg;
            if (r < cfg::N && m < cfg::N) {
                lg = s[t][i] * scale
                   + rpb[rpi[r * cfg::N + m] * cfg::NH + head]
                   + mk[r * cfg::N + m];
            } else {
                lg = -1e30f;   // pad rows/cols vanish under softmax
            }
            s[t][i] = lg;
        }

    // Row softmax: a row's 64 cols live in 16 lanes of this half-group across
    // the 4 tiles -> in-lane max over t, then shfl_xor 1/2/4/8.
#pragma unroll
    for (int i = 0; i < 8; ++i) {
        float mx = fmaxf(fmaxf(s[0][i], s[1][i]), fmaxf(s[2][i], s[3][i]));
#pragma unroll
        for (int d = 1; d < 16; d <<= 1) mx = fmaxf(mx, __shfl_xor(mx, d, 32));
        float sum = 0.f;
#pragma unroll
        for (int t = 0; t < 4; ++t) {
            float e = __expf(s[t][i] - mx);
            s[t][i] = e;
            sum += e;
        }
#pragma unroll
        for (int d = 1; d < 16; d <<= 1) sum += __shfl_xor(sum, d, 32);
        float inv = 1.f / sum;
#pragma unroll
        for (int t = 0; t < 4; ++t) s[t][i] *= inv;
    }

    // P -> LDS (bf16) for the second WMMA pass
#pragma unroll
    for (int t = 0; t < 4; ++t)
#pragma unroll
        for (int i = 0; i < 8; ++i) {
            int r = wv * 16 + 8 * half + i;
            int m = t * 16 + cl;
            Ps[r * 72 + m] = (__bf16)s[t][i];
        }
    __syncthreads();

    // O = P(64x64) @ V(64x32): 2 K-chunks x 2 N-tiles per wave
    v8f o[2] = {};
#pragma unroll
    for (int kc = 0; kc < 2; ++kc) {
        v16bf a = frag_a(Ps, 72, wv * 16, kc * 32);
#pragma unroll
        for (int t = 0; t < 2; ++t) {
            v16bf b = frag_bT(Vt, 72, kc * 32, t * 16);
            o[t] = wmma_bf16(a, b, o[t]);
        }
    }
#pragma unroll
    for (int t = 0; t < 2; ++t)
#pragma unroll
        for (int i = 0; i < 8; ++i) {
            int r = wv * 16 + 8 * half + i;
            if (r < cfg::N) {
                aout[(size_t)win * cfg::N * cfg::C + (size_t)r * cfg::C +
                     head * cfg::HD + t * 16 + cl] = (__bf16)o[t][i];
            }
        }
}

// proj: aout(2048x49x384) @ proj_w(384x384) + b, window-reverse + roll(+3,+3),
// add residual x -> out fp32 (full 100352x384 coverage).
__global__ __launch_bounds__(128)
void proj_merge_kernel(const __bf16* __restrict__ aout,
                       const __bf16* __restrict__ wpT,   // (384 x 384)
                       const float* __restrict__ pb,
                       const float* __restrict__ xin,
                       float* __restrict__ out) {
    int win  = blockIdx.y;
    int col0 = blockIdx.x * 64;
    __shared__ __align__(16) __bf16 As[2 * cfg::TILE];
    __shared__ __align__(16) __bf16 Bs[2 * cfg::TILE];
    int tid = threadIdx.x, wv = tid >> 5, lane = tid & 31;
    int half = lane >> 4, cl = lane & 15;
    v8f acc[4] = {};
    gemm_mainloop<cfg::C, cfg::N>(As, Bs,
                                  aout + (size_t)win * cfg::N * cfg::C, cfg::C,
                                  wpT + (size_t)col0 * cfg::C, cfg::C,
                                  tid, wv, acc);
    int bidx = win >> 6, wrem = win & 63;
    int wi = wrem >> 3, wj = wrem & 7;
#pragma unroll
    for (int t = 0; t < 4; ++t)
#pragma unroll
        for (int i = 0; i < 8; ++i) {
            int r = wv * 16 + 8 * half + i;
            if (r < cfg::N) {
                int hh = wi * cfg::WS + r / cfg::WS;
                int ww = wj * cfg::WS + r % cfg::WS;
                int dh = (hh + cfg::SHIFT) % cfg::HS;
                int dw = (ww + cfg::SHIFT) % cfg::WSS;
                size_t tok = (size_t)bidx * 3136 + dh * 56 + dw;
                int col = col0 + t * 16 + cl;
                out[tok * cfg::C + col] =
                    xin[tok * cfg::C + col] + acc[t][i] + pb[col];
            }
        }
}

// fc1 + exact GELU: hln(100352x384) @ fc1_w(384x1536) + b -> h2 bf16
__global__ __launch_bounds__(128)
void fc1_gelu_kernel(const __bf16* __restrict__ hln,
                     const __bf16* __restrict__ w1T,   // (1536 x 384)
                     const float* __restrict__ b1,
                     __bf16* __restrict__ h2) {
    int row0 = blockIdx.y * 64;
    int col0 = blockIdx.x * 64;
    __shared__ __align__(16) __bf16 As[2 * cfg::TILE];
    __shared__ __align__(16) __bf16 Bs[2 * cfg::TILE];
    int tid = threadIdx.x, wv = tid >> 5, lane = tid & 31;
    int half = lane >> 4, cl = lane & 15;
    v8f acc[4] = {};
    gemm_mainloop<cfg::C, 64>(As, Bs,
                              hln + (size_t)row0 * cfg::C, cfg::C,
                              w1T + (size_t)col0 * cfg::C, cfg::C,
                              tid, wv, acc);
#pragma unroll
    for (int t = 0; t < 4; ++t)
#pragma unroll
        for (int i = 0; i < 8; ++i) {
            int r = row0 + wv * 16 + 8 * half + i;
            int col = col0 + t * 16 + cl;
            float v = acc[t][i] + b1[col];
            float g = 0.5f * v * (1.f + erff(v * 0.70710678118654752f));
            h2[(size_t)r * cfg::MLP + col] = (__bf16)g;
        }
}

// fc2 + residual: h2(100352x1536) @ fc2_w(1536x384) + b + out -> out fp32
__global__ __launch_bounds__(128)
void fc2_res_kernel(const __bf16* __restrict__ h2,
                    const __bf16* __restrict__ w2T,   // (384 x 1536)
                    const float* __restrict__ b2,
                    float* __restrict__ out) {
    int row0 = blockIdx.y * 64;
    int col0 = blockIdx.x * 64;
    __shared__ __align__(16) __bf16 As[2 * cfg::TILE];
    __shared__ __align__(16) __bf16 Bs[2 * cfg::TILE];
    int tid = threadIdx.x, wv = tid >> 5, lane = tid & 31;
    int half = lane >> 4, cl = lane & 15;
    v8f acc[4] = {};
    gemm_mainloop<cfg::MLP, 64>(As, Bs,
                                h2 + (size_t)row0 * cfg::MLP, cfg::MLP,
                                w2T + (size_t)col0 * cfg::MLP, cfg::MLP,
                                tid, wv, acc);
#pragma unroll
    for (int t = 0; t < 4; ++t)
#pragma unroll
        for (int i = 0; i < 8; ++i) {
            int r = row0 + wv * 16 + 8 * half + i;
            int col = col0 + t * 16 + cl;
            size_t idx = (size_t)r * cfg::C + col;
            out[idx] = out[idx] + acc[t][i] + b2[col];
        }
}

// ------------------------------- launcher ----------------------------------

extern "C" void kernel_launch(void* const* d_in, const int* in_sizes, int n_in,
                              void* d_out, int out_size, void* d_ws, size_t ws_size,
                              hipStream_t stream) {
    (void)in_sizes; (void)n_in; (void)out_size; (void)ws_size;

    const float* x     = (const float*)d_in[0];
    const float* maskm = (const float*)d_in[1];
    const int*   rpi   = (const int*)d_in[2];
    const float* n1w   = (const float*)d_in[3];
    const float* n1b   = (const float*)d_in[4];
    const float* qkvw  = (const float*)d_in[5];
    const float* qkvb  = (const float*)d_in[6];
    const float* rpbt  = (const float*)d_in[7];
    const float* projw = (const float*)d_in[8];
    const float* projb = (const float*)d_in[9];
    const float* n2w   = (const float*)d_in[10];
    const float* n2b   = (const float*)d_in[11];
    const float* f1w   = (const float*)d_in[12];
    const float* f1b   = (const float*)d_in[13];
    const float* f2w   = (const float*)d_in[14];
    const float* f2b   = (const float*)d_in[15];
    float* out = (float*)d_out;

    // Workspace layout (bf16 elements, ~466 MB total):
    //   weightsT | xw (38.5M) | aout/hln (38.5M) | big: qkvbuf(115.6M)/h2(154.1M)
    __bf16* ws = (__bf16*)d_ws;
    size_t o = 0;
    __bf16* wqT = ws + o; o += (size_t)1152 * 384;
    __bf16* wpT = ws + o; o += (size_t)384 * 384;
    __bf16* w1T = ws + o; o += (size_t)1536 * 384;
    __bf16* w2T = ws + o; o += (size_t)384 * 1536;
    o = (o + 127) & ~(size_t)127;
    __bf16* xw   = ws + o; o += (size_t)cfg::NWIN * cfg::N * cfg::C;   // 38.5M
    __bf16* aout = ws + o; o += (size_t)cfg::NWIN * cfg::N * cfg::C;   // 38.5M
    __bf16* big  = ws + o;                                             // 154.1M
    __bf16* qkvbuf = big;          // dead after attention
    __bf16* h2     = big;          // reuses qkvbuf region
    __bf16* hln    = aout;         // reuses attention output region

    // 0) weights -> bf16, transposed to (N x K) row-major
    cvt_transpose_kernel<<<(384 * 1152 + 255) / 256, 256, 0, stream>>>(qkvw, wqT, 384, 1152);
    cvt_transpose_kernel<<<(384 * 384  + 255) / 256, 256, 0, stream>>>(projw, wpT, 384, 384);
    cvt_transpose_kernel<<<(384 * 1536 + 255) / 256, 256, 0, stream>>>(f1w, w1T, 384, 1536);
    cvt_transpose_kernel<<<(1536 * 384 + 255) / 256, 256, 0, stream>>>(f2w, w2T, 1536, 384);

    // 1) LN1 + shift + window partition (bf16)
    ln1_window_kernel<<<cfg::TOK / 8, 256, 0, stream>>>(x, n1w, n1b, xw);

    // 2) QKV projection
    qkv_gemm_kernel<<<dim3(1152 / 64, cfg::NWIN), 128, 0, stream>>>(xw, wqT, qkvb, qkvbuf);

    // 3) windowed attention (per head x window)
    attn_kernel<<<dim3(cfg::NH, cfg::NWIN), 128, 0, stream>>>(qkvbuf, rpbt, rpi, maskm, aout);

    // 4) proj + window reverse + roll back + residual -> out (fp32)
    proj_merge_kernel<<<dim3(cfg::C / 64, cfg::NWIN), 128, 0, stream>>>(aout, wpT, projb, x, out);

    // 5) LN2 (bf16)
    ln2_kernel<<<cfg::TOK / 8, 256, 0, stream>>>(out, n2w, n2b, hln);

    // 6) fc1 + GELU
    fc1_gelu_kernel<<<dim3(cfg::MLP / 64, cfg::TOK / 64), 128, 0, stream>>>(hln, w1T, f1b, h2);

    // 7) fc2 + residual -> out
    fc2_res_kernel<<<dim3(cfg::C / 64, cfg::TOK / 64), 128, 0, stream>>>(h2, w2T, f2b, out);
}
